// ParallelExperts_69191923138886
// MI455X (gfx1250) — compile-verified, matrix operand order checked by reading, not run on hardware
//
#include <hip/hip_runtime.h>

// ---------------- problem constants (fixed by the reference) ----------------
#define N_EXPERTS 8
#define IN_DIM    1024
#define OUT_DIM   1024
#define N_TOKENS  16384
#define TOPK      2
#define S_SLOTS   (N_TOKENS * TOPK)   // 32768 slots

// ---------------- tiling ----------------
#define BM   128        // slot rows per block
#define BN   128        // output cols per block
#define BK   32         // K step == WMMA K for 16x16x32 bf16
#define KT   (IN_DIM / BK)   // 32 k-steps
#define LDSK 40         // padded K stride (bf16 elems): 80B rows, 16B-aligned chunks
#define NTHREADS 256    // 8 wave32s

typedef __attribute__((ext_vector_type(16))) __bf16 v16bf;
typedef __attribute__((ext_vector_type(8)))  float  v8f;

union FragBF { v16bf v; uint4 q[2]; };
union H4     { __bf16 h[4]; uint2 u; };

// ---------------------------------------------------------------------------
// CDNA5 async global->LDS copy (ASYNCcnt-tracked), per-lane 16B.
// vDst = LDS byte address (low 32 bits of a generic shared pointer), per-lane.
// ---------------------------------------------------------------------------
__device__ __forceinline__ void async_ld_b128(unsigned int lds_addr, const void* gaddr) {
    asm volatile("global_load_async_to_lds_b128 %0, %1, off"
                 :: "v"(lds_addr), "v"(gaddr)
                 : "memory");
}
__device__ __forceinline__ unsigned int lds_addr_of(const void* p) {
    return (unsigned int)(unsigned long long)p;   // LDS flat addr: low 32 bits = LDS offset
}

// ===========================================================================
// Phase 1: split f32 -> bf16 hi/lo planes (hi = RNE(x), lo = RNE(x - hi))
// ===========================================================================
__global__ __launch_bounds__(256)
void split_f32_bf16_kernel(const float* __restrict__ src,
                           unsigned short* __restrict__ hi,
                           unsigned short* __restrict__ lo,
                           int n4)   // number of float4 groups
{
    const int i = blockIdx.x * blockDim.x + threadIdx.x;
    if (i >= n4) return;
    const float4 v = ((const float4*)src)[i];
    H4 hh, ll;
#pragma unroll
    for (int j = 0; j < 4; ++j) {
        const float f = (&v.x)[j];
        const __bf16 h = (__bf16)f;
        hh.h[j] = h;
        ll.h[j] = (__bf16)(f - (float)h);
    }
    ((uint2*)hi)[i] = hh.u;
    ((uint2*)lo)[i] = ll.u;
}

// ===========================================================================
// Phase 2: slot GEMM, bf16x3 WMMA, async double-buffered LDS staging
// ===========================================================================
__global__ __launch_bounds__(NTHREADS)
void moe_gemm_async_kernel(const unsigned short* __restrict__ xh,
                           const unsigned short* __restrict__ xl,
                           const unsigned short* __restrict__ wh,
                           const unsigned short* __restrict__ wl,
                           const float* __restrict__ gates,
                           const int*   __restrict__ sorted_expert_idxs,
                           const int*   __restrict__ sorted_scattered_idxs,
                           float*       __restrict__ out)
{
    __shared__ unsigned short smA[2][2][BM * LDSK];   // [buf][hi/lo][row*LDSK+k]
    __shared__ unsigned short smB[2][2][BN * LDSK];
    __shared__ int s_tok[BM];
    __shared__ int s_exp[BM];

    const int tid  = threadIdx.x;
    const int lane = tid & 31;
    const int wave = tid >> 5;
    const int wm   = wave & 3;    // rows wm*32 .. +31
    const int wn   = wave >> 2;   // cols wn*64 .. +63

    const int slot_base = blockIdx.x * BM;
    const int n_base    = blockIdx.y * BN;

    // ---- slot metadata ----
    for (int i = tid; i < BM; i += NTHREADS) {
        const int slot = slot_base + i;
        s_tok[i] = sorted_scattered_idxs[slot] / TOPK;
        s_exp[i] = sorted_expert_idxs[slot];
    }
    __syncthreads();
    const int e_first = s_exp[0];
    const int e_last  = s_exp[BM - 1];
    const int P       = e_last - e_first + 1;   // expert passes (usually 1)
    const int T       = P * KT;                 // total k-tiles

    // ---- accumulators ----
    v8f acc[2][4];
#pragma unroll
    for (int mi = 0; mi < 2; ++mi)
#pragma unroll
        for (int ni = 0; ni < 4; ++ni)
#pragma unroll
            for (int q = 0; q < 8; ++q) acc[mi][ni][q] = 0.0f;

    const int khalf = (lane >> 4) * 8;
    const int r16   = lane & 15;

    // per-thread staging coordinates: 2 iterations x (row, 16B chunk)
    const int row0 = (tid + 0 * NTHREADS) >> 2, c0 = (tid + 0 * NTHREADS) & 3;
    const int row1 = (tid + 1 * NTHREADS) >> 2, c1 = (tid + 1 * NTHREADS) & 3;

    // issue 8 async b128 copies (A hi/lo gather rows + B hi/lo weight rows)
    auto issue_tile = [&](int buf, int e, int kb) {
        const unsigned int aH = lds_addr_of(&smA[buf][0][0]);
        const unsigned int aL = lds_addr_of(&smA[buf][1][0]);
        const unsigned int bH = lds_addr_of(&smB[buf][0][0]);
        const unsigned int bL = lds_addr_of(&smB[buf][1][0]);
        {
            const unsigned int lo_off = (unsigned int)(row0 * LDSK + c0 * 8) * 2u;
            const size_t ga = (size_t)s_tok[row0] * IN_DIM + kb + c0 * 8;
            const size_t gb = ((size_t)e * OUT_DIM + (n_base + row0)) * IN_DIM + kb + c0 * 8;
            async_ld_b128(aH + lo_off, xh + ga);
            async_ld_b128(aL + lo_off, xl + ga);
            async_ld_b128(bH + lo_off, wh + gb);
            async_ld_b128(bL + lo_off, wl + gb);
        }
        {
            const unsigned int lo_off = (unsigned int)(row1 * LDSK + c1 * 8) * 2u;
            const size_t ga = (size_t)s_tok[row1] * IN_DIM + kb + c1 * 8;
            const size_t gb = ((size_t)e * OUT_DIM + (n_base + row1)) * IN_DIM + kb + c1 * 8;
            async_ld_b128(aH + lo_off, xh + ga);
            async_ld_b128(aL + lo_off, xl + ga);
            async_ld_b128(bH + lo_off, wh + gb);
            async_ld_b128(bL + lo_off, wl + gb);
        }
    };

    // ---- pipeline prologue: tile 0 in flight ----
    issue_tile(0, e_first, 0);

    for (int t = 0; t < T; ++t) {
        const int buf = t & 1;
        const int e   = e_first + t / KT;
        const int kb  = (t % KT) * BK;

        __syncthreads();   // (a) all waves done with compute(t-1) -> buf[(t+1)&1] free
        if (t + 1 < T) {
            issue_tile((t + 1) & 1, e_first + (t + 1) / KT, ((t + 1) % KT) * BK);
            asm volatile("s_wait_asynccnt 0x8" ::: "memory");   // tile t's 8 copies done
        } else {
            asm volatile("s_wait_asynccnt 0x0" ::: "memory");
        }
        __syncthreads();   // (b) tile t visible to all waves

        // expert-boundary masking (rare): zero A rows not owned by expert e
        if (P > 1) {
            if (tid < BM && s_exp[tid] != e) {
                const uint4 z = make_uint4(0u, 0u, 0u, 0u);
                unsigned short* rh = &smA[buf][0][tid * LDSK];
                unsigned short* rl = &smA[buf][1][tid * LDSK];
#pragma unroll
                for (int c = 0; c < 5; ++c) {   // 5 x 16B = 80B = LDSK halves
                    *(uint4*)(rh + c * 8) = z;
                    *(uint4*)(rl + c * 8) = z;
                }
            }
            __syncthreads();
        }

        // ---- fragments (ds_load_b128 pairs) ----
        FragBF ah[2], al[2], bh[4], bl[4];
#pragma unroll
        for (int mi = 0; mi < 2; ++mi) {
            const unsigned short* ph = &smA[buf][0][(wm * 32 + mi * 16 + r16) * LDSK + khalf];
            const unsigned short* pl = &smA[buf][1][(wm * 32 + mi * 16 + r16) * LDSK + khalf];
            ah[mi].q[0] = *(const uint4*)(ph);
            ah[mi].q[1] = *(const uint4*)(ph + 16);
            al[mi].q[0] = *(const uint4*)(pl);
            al[mi].q[1] = *(const uint4*)(pl + 16);
        }
#pragma unroll
        for (int ni = 0; ni < 4; ++ni) {
            const unsigned short* ph = &smB[buf][0][(wn * 64 + ni * 16 + r16) * LDSK + khalf];
            const unsigned short* pl = &smB[buf][1][(wn * 64 + ni * 16 + r16) * LDSK + khalf];
            bh[ni].q[0] = *(const uint4*)(ph);
            bh[ni].q[1] = *(const uint4*)(ph + 16);
            bl[ni].q[0] = *(const uint4*)(pl);
            bl[ni].q[1] = *(const uint4*)(pl + 16);
        }

        // ---- bf16x3: C += Ah*Bh + Ah*Bl + Al*Bh ----
#pragma unroll
        for (int mi = 0; mi < 2; ++mi)
#pragma unroll
            for (int ni = 0; ni < 4; ++ni) {
                v8f c = acc[mi][ni];
                c = __builtin_amdgcn_wmma_f32_16x16x32_bf16(
                        false, ah[mi].v, false, bh[ni].v, (short)0, c, false, false);
                c = __builtin_amdgcn_wmma_f32_16x16x32_bf16(
                        false, ah[mi].v, false, bl[ni].v, (short)0, c, false, false);
                c = __builtin_amdgcn_wmma_f32_16x16x32_bf16(
                        false, al[mi].v, false, bh[ni].v, (short)0, c, false, false);
                acc[mi][ni] = c;
            }
    }

    // ---- fused gate combine (in-lane: slot rows 2t / 2t+1 are adjacent C VGPRs) ----
    const int row_half = (lane >> 4) * 8;
#pragma unroll
    for (int mi = 0; mi < 2; ++mi) {
        const int rbase = wm * 32 + mi * 16;
#pragma unroll
        for (int j = 0; j < 4; ++j) {
            const int row = rbase + 2 * j + row_half;
            const int t   = (slot_base + row) >> 1;
            const float2 g = *(const float2*)(gates + (size_t)t * TOPK);
#pragma unroll
            for (int ni = 0; ni < 4; ++ni) {
                const int col = n_base + wn * 64 + ni * 16 + r16;
                out[(size_t)t * OUT_DIM + col] =
                    g.x * acc[mi][ni][2 * j] + g.y * acc[mi][ni][2 * j + 1];
            }
        }
    }
}

// ===========================================================================
// Fallback (round-1 kernel): fused in-loop f32->bf16 split, no workspace
// ===========================================================================
__global__ __launch_bounds__(NTHREADS)
void moe_slot_gemm_bf16x3_kernel(const float* __restrict__ inputs,
                                 const float* __restrict__ weight,
                                 const float* __restrict__ gates,
                                 const int*   __restrict__ sorted_expert_idxs,
                                 const int*   __restrict__ sorted_scattered_idxs,
                                 float*       __restrict__ out)
{
    __shared__ __bf16 smAh[BM * LDSK];
    __shared__ __bf16 smAl[BM * LDSK];
    __shared__ __bf16 smBh[BN * LDSK];
    __shared__ __bf16 smBl[BN * LDSK];
    __shared__ int s_tok[BM];
    __shared__ int s_exp[BM];

    const int tid  = threadIdx.x;
    const int lane = tid & 31;
    const int wave = tid >> 5;
    const int wm   = wave & 3;
    const int wn   = wave >> 2;
    const int slot_base = blockIdx.x * BM;
    const int n_base    = blockIdx.y * BN;

    for (int i = tid; i < BM; i += NTHREADS) {
        const int slot = slot_base + i;
        s_tok[i] = sorted_scattered_idxs[slot] / TOPK;
        s_exp[i] = sorted_expert_idxs[slot];
    }
    __syncthreads();
    const int e_first = s_exp[0];
    const int e_last  = s_exp[BM - 1];

    v8f acc[2][4];
#pragma unroll
    for (int mi = 0; mi < 2; ++mi)
#pragma unroll
        for (int ni = 0; ni < 4; ++ni)
#pragma unroll
            for (int q = 0; q < 8; ++q) acc[mi][ni][q] = 0.0f;

    const int khalf = (lane >> 4) * 8;
    const int r16   = lane & 15;

    for (int e = e_first; e <= e_last; ++e) {
        const float* wbase = weight + (size_t)e * (size_t)(OUT_DIM * IN_DIM);
        for (int kb = 0; kb < IN_DIM; kb += BK) {
            __syncthreads();
#pragma unroll
            for (int it = 0; it < 4; ++it) {
                const int idx = tid + it * NTHREADS;
                const int row = idx >> 3;
                const int c4  = idx & 7;
                const float mask = (s_exp[row] == e) ? 1.0f : 0.0f;
                const float* src = inputs + (size_t)s_tok[row] * IN_DIM + kb + c4 * 4;
                const float4 v = *(const float4*)src;
                H4 hh, ll;
#pragma unroll
                for (int j = 0; j < 4; ++j) {
                    const float f = (&v.x)[j] * mask;
                    const __bf16 h = (__bf16)f;
                    hh.h[j] = h;
                    ll.h[j] = (__bf16)(f - (float)h);
                }
                *(uint2*)(smAh + row * LDSK + c4 * 4) = hh.u;
                *(uint2*)(smAl + row * LDSK + c4 * 4) = ll.u;
            }
#pragma unroll
            for (int it = 0; it < 4; ++it) {
                const int idx = tid + it * NTHREADS;
                const int row = idx >> 3;
                const int c4  = idx & 7;
                const float* src = wbase + (size_t)(n_base + row) * IN_DIM + kb + c4 * 4;
                const float4 v = *(const float4*)src;
                H4 hh, ll;
#pragma unroll
                for (int j = 0; j < 4; ++j) {
                    const float f = (&v.x)[j];
                    const __bf16 h = (__bf16)f;
                    hh.h[j] = h;
                    ll.h[j] = (__bf16)(f - (float)h);
                }
                *(uint2*)(smBh + row * LDSK + c4 * 4) = hh.u;
                *(uint2*)(smBl + row * LDSK + c4 * 4) = ll.u;
            }
            __syncthreads();

            FragBF ah[2], al[2], bh[4], bl[4];
#pragma unroll
            for (int mi = 0; mi < 2; ++mi) {
                const __bf16* ph = smAh + (wm * 32 + mi * 16 + r16) * LDSK + khalf;
                const __bf16* pl = smAl + (wm * 32 + mi * 16 + r16) * LDSK + khalf;
                ah[mi].q[0] = *(const uint4*)(ph);
                ah[mi].q[1] = *(const uint4*)(ph + 16);
                al[mi].q[0] = *(const uint4*)(pl);
                al[mi].q[1] = *(const uint4*)(pl + 16);
            }
#pragma unroll
            for (int ni = 0; ni < 4; ++ni) {
                const __bf16* ph = smBh + (wn * 64 + ni * 16 + r16) * LDSK + khalf;
                const __bf16* pl = smBl + (wn * 64 + ni * 16 + r16) * LDSK + khalf;
                bh[ni].q[0] = *(const uint4*)(ph);
                bh[ni].q[1] = *(const uint4*)(ph + 16);
                bl[ni].q[0] = *(const uint4*)(pl);
                bl[ni].q[1] = *(const uint4*)(pl + 16);
            }
#pragma unroll
            for (int mi = 0; mi < 2; ++mi)
#pragma unroll
                for (int ni = 0; ni < 4; ++ni) {
                    v8f c = acc[mi][ni];
                    c = __builtin_amdgcn_wmma_f32_16x16x32_bf16(
                            false, ah[mi].v, false, bh[ni].v, (short)0, c, false, false);
                    c = __builtin_amdgcn_wmma_f32_16x16x32_bf16(
                            false, ah[mi].v, false, bl[ni].v, (short)0, c, false, false);
                    c = __builtin_amdgcn_wmma_f32_16x16x32_bf16(
                            false, al[mi].v, false, bh[ni].v, (short)0, c, false, false);
                    acc[mi][ni] = c;
                }
        }
    }

    const int row_half = (lane >> 4) * 8;
#pragma unroll
    for (int mi = 0; mi < 2; ++mi) {
        const int rbase = wm * 32 + mi * 16;
#pragma unroll
        for (int j = 0; j < 4; ++j) {
            const int row = rbase + 2 * j + row_half;
            const int t   = (slot_base + row) >> 1;
            const float2 g = *(const float2*)(gates + (size_t)t * TOPK);
#pragma unroll
            for (int ni = 0; ni < 4; ++ni) {
                const int col = n_base + wn * 64 + ni * 16 + r16;
                out[(size_t)t * OUT_DIM + col] =
                    g.x * acc[mi][ni][2 * j] + g.y * acc[mi][ni][2 * j + 1];
            }
        }
    }
}

// ---------------------------------------------------------------------------
extern "C" void kernel_launch(void* const* d_in, const int* in_sizes, int n_in,
                              void* d_out, int out_size, void* d_ws, size_t ws_size,
                              hipStream_t stream) {
    (void)in_sizes; (void)n_in; (void)out_size;
    const float* inputs = (const float*)d_in[0];
    const float* weight = (const float*)d_in[1];
    const float* gates  = (const float*)d_in[2];
    const int* sorted_expert_idxs    = (const int*)d_in[4];
    const int* sorted_scattered_idxs = (const int*)d_in[5];
    float* out = (float*)d_out;

    const size_t XP = (size_t)N_TOKENS * IN_DIM;              // input plane elems
    const size_t WP = (size_t)N_EXPERTS * OUT_DIM * IN_DIM;   // weight plane elems
    const size_t need = 2 * (XP + WP) * sizeof(unsigned short);

    dim3 grid(S_SLOTS / BM, OUT_DIM / BN);   // 256 x 8

    if (ws_size >= need) {
        unsigned short* xh = (unsigned short*)d_ws;
        unsigned short* xl = xh + XP;
        unsigned short* wh = xl + XP;
        unsigned short* wl = wh + WP;

        const int xn4 = (int)(XP / 4);
        const int wn4 = (int)(WP / 4);
        split_f32_bf16_kernel<<<(xn4 + 255) / 256, 256, 0, stream>>>(inputs, xh, xl, xn4);
        split_f32_bf16_kernel<<<(wn4 + 255) / 256, 256, 0, stream>>>(weight, wh, wl, wn4);

        moe_gemm_async_kernel<<<grid, NTHREADS, 0, stream>>>(
            xh, xl, wh, wl, gates, sorted_expert_idxs, sorted_scattered_idxs, out);
    } else {
        moe_slot_gemm_bf16x3_kernel<<<grid, NTHREADS, 0, stream>>>(
            inputs, weight, gates, sorted_expert_idxs, sorted_scattered_idxs, out);
    }
}
